// TFIdeficsPerceiverResampler_13451837571518
// MI455X (gfx1250) — matrix operand order repaired and checked by reading, not compiled, not run on hardware
//
#include <hip/hip_runtime.h>

// ---------------- problem constants ----------------
#define DD     1280
#define NHEAD  16
#define HDIM   96
#define NL     64
#define INTERN 5120
#define BBATCH 32
#define SSEQ   2304
#define TTOT   2368          // S + L
#define NHHD   1536          // H * HD
#define EPSL   1e-5f

// ---------------- vector types ----------------
typedef unsigned int u32x4  __attribute__((ext_vector_type(4)));
typedef float        f32x8  __attribute__((ext_vector_type(8)));
typedef __bf16       bf16x16 __attribute__((ext_vector_type(16)));
typedef int          v4i    __attribute__((vector_size(16)));   // matches builtin param

union Frag { bf16x16 m; u32x4 u[2]; };

// ---------------- CDNA5 async global->LDS path (ASYNCcnt) ----------------
#if defined(__AMDGCN__) && __has_builtin(__builtin_amdgcn_global_load_async_to_lds_b128) && \
    __has_builtin(__builtin_amdgcn_s_wait_asynccnt)
#define HAVE_ASYNC_LDS 1
typedef __attribute__((address_space(1))) v4i* gptr_v4i;   // global (AS1)
typedef __attribute__((address_space(3))) v4i* lptr_v4i;   // LDS (AS3)
#else
#define HAVE_ASYNC_LDS 0
#endif

static __device__ __forceinline__ void cp16_async(unsigned short* dstLds, const unsigned short* srcGlb) {
#if HAVE_ASYNC_LDS
    __builtin_amdgcn_global_load_async_to_lds_b128(
        (gptr_v4i)(void*)srcGlb,
        (lptr_v4i)(void*)dstLds, 0, 0);
#else
    *(u32x4*)dstLds = *(const u32x4*)srcGlb;
#endif
}
static __device__ __forceinline__ void wait_async0() {
#if HAVE_ASYNC_LDS
    __builtin_amdgcn_s_wait_asynccnt(0);
#endif
}

static __device__ __forceinline__ unsigned short f2bf(float f) {
    unsigned u = __float_as_uint(f);
    u = (u + 0x7fffu + ((u >> 16) & 1u)) >> 16;   // round-to-nearest-even
    return (unsigned short)u;
}
static __device__ __forceinline__ float bf2f(unsigned short h) {
    return __uint_as_float(((unsigned)h) << 16);
}
static __device__ __forceinline__ f32x8 zero8() {
    f32x8 z;
#pragma unroll
    for (int i = 0; i < 8; i++) z[i] = 0.f;
    return z;
}
static __device__ __forceinline__ f32x8 wmma_bf(const Frag& a, const Frag& b, f32x8 c) {
    return __builtin_amdgcn_wmma_f32_16x16x32_bf16(false, a.m, false, b.m, (short)0, c, false, false);
}

// =====================================================================
// Row LayerNorm: f32 [R,C] -> bf16 (or f32) [R,C].  One wave per row.
// g/b may be null (pure normalize).
// =====================================================================
__global__ __launch_bounds__(256) void ln_rows_kernel(
    const float* __restrict__ x, const float* __restrict__ g, const float* __restrict__ b,
    void* __restrict__ out, int R, int C, int outF32)
{
    int wv = threadIdx.x >> 5, lane = threadIdx.x & 31;
    int row = blockIdx.x * 8 + wv;
    if (row >= R) return;
    const float* xr = x + (size_t)row * C;
    float s = 0.f, s2 = 0.f;
    for (int i = lane; i < C; i += 32) { float v = xr[i]; s += v; s2 += v * v; }
#pragma unroll
    for (int o = 1; o < 32; o <<= 1) { s += __shfl_xor(s, o, 32); s2 += __shfl_xor(s2, o, 32); }
    float mean = s / C;
    float var  = s2 / C - mean * mean;
    float rs   = rsqrtf(var + EPSL);
    for (int i = lane; i < C; i += 32) {
        float v = (xr[i] - mean) * rs;
        if (g) v = v * g[i] + b[i];
        if (outF32) ((float*)out)[(size_t)row * C + i] = v;
        else ((unsigned short*)out)[(size_t)row * C + i] = f2bf(v);
    }
}

// =====================================================================
// Per-head LN over HD=96 (QK-norm).  Input bf16 [Rtotal,1536] row-major.
// Output bf16 scattered to [b, h, t, 96] with t = rr%Trows + toff, T-extent TT.
// One wave per (row, head); 3 elements per lane.
// =====================================================================
__global__ __launch_bounds__(256) void ln_head_kernel(
    const unsigned short* __restrict__ in, const float* __restrict__ g, const float* __restrict__ b,
    unsigned short* __restrict__ out, int Rtotal, int Trows, int TT, int toff, float postScale)
{
    int wv = threadIdx.x >> 5, lane = threadIdx.x & 31;
    long gw = (long)blockIdx.x * 8 + wv;
    if (gw >= (long)Rtotal * NHEAD) return;
    long rr = gw >> 4; int h = (int)(gw & 15);
    const unsigned short* xr = in + rr * NHHD + h * HDIM;
    float v0 = bf2f(xr[lane]), v1 = bf2f(xr[lane + 32]), v2 = bf2f(xr[lane + 64]);
    float s = v0 + v1 + v2, s2 = v0 * v0 + v1 * v1 + v2 * v2;
#pragma unroll
    for (int o = 1; o < 32; o <<= 1) { s += __shfl_xor(s, o, 32); s2 += __shfl_xor(s2, o, 32); }
    float mean = s / HDIM;
    float var  = s2 / HDIM - mean * mean;
    float rs   = rsqrtf(var + EPSL);
    int bb = (int)(rr / Trows);
    int t  = (int)(rr % Trows) + toff;
    unsigned short* orow = out + (((size_t)bb * NHEAD + h) * TT + t) * HDIM;
    orow[lane]      = f2bf(((v0 - mean) * rs * g[lane]      + b[lane])      * postScale);
    orow[lane + 32] = f2bf(((v1 - mean) * rs * g[lane + 32] + b[lane + 32]) * postScale);
    orow[lane + 64] = f2bf(((v2 - mean) * rs * g[lane + 64] + b[lane + 64]) * postScale);
}

// =====================================================================
// Transpose (+ optional per-K row scale) : W f32 [K,N] -> WT bf16 [N,K]
// =====================================================================
__global__ __launch_bounds__(256) void tsp_kernel(
    const float* __restrict__ W, const float* __restrict__ g,
    unsigned short* __restrict__ WT, int K, int N)
{
    __shared__ float tile[32][33];
    int n0 = blockIdx.x * 32, k0 = blockIdx.y * 32;
    int tx = threadIdx.x, ty = threadIdx.y;          // 32 x 8
#pragma unroll
    for (int i = 0; i < 4; i++) {
        int r = ty + i * 8;
        float v = W[(size_t)(k0 + r) * N + n0 + tx];
        if (g) v *= g[k0 + r];
        tile[r][tx] = v;
    }
    __syncthreads();
#pragma unroll
    for (int i = 0; i < 4; i++) {
        int r = ty + i * 8;
        WT[(size_t)(n0 + r) * K + k0 + tx] = f2bf(tile[tx][r]);
    }
}

// bias[n] = sum_k bvec[k] * W[k,n]    (LN-shift folded through weight)
__global__ __launch_bounds__(256) void gemv_bias_kernel(
    const float* __restrict__ bvec, const float* __restrict__ W,
    float* __restrict__ bias, int K, int N)
{
    int n = blockIdx.x * 256 + threadIdx.x;
    if (n >= N) return;
    float acc = 0.f;
    for (int k = 0; k < K; k++) acc += bvec[k] * W[(size_t)k * N + n];
    bias[n] = acc;
}

// lat[b,l,d] = latents[l,d]
__global__ void bcast_lat_kernel(const float* __restrict__ latents, float* __restrict__ lat, int total, int per)
{
    int i = blockIdx.x * 256 + threadIdx.x;
    if (i < total) lat[i] = latents[i % per];
}

// =====================================================================
// Tiled bf16 GEMM with WMMA:  C[M,N] = A[M,K](bf16) * WT[N,K](bf16)^T
//   + optional bias[N], relu, residual(f32).
// 256 threads = 8 waves (4 M-groups x 2 N-groups); tile 128x128, BK=32.
// Async global->LDS double-buffered staging (ASYNCcnt) on gfx1250.
// Epilogue modes:
//   0: bf16, row remapped: row' = (m/Trows)*TT + m%Trows + toff   (out [_,N])
//   1: f32 row-major [M,N], += residual
//   2: bf16 V-scatter: out[((b*16+h)*96+d)*TT + t]   (d-major for attention)
// All of M,N are multiples of 128 and K of 32 in this problem: no bounds.
// =====================================================================
__global__ __launch_bounds__(256) void gemm_bf16_kernel(
    const unsigned short* __restrict__ A, const unsigned short* __restrict__ W,
    const float* __restrict__ bias, const float* __restrict__ resid,
    void* __restrict__ out, int M, int N, int K,
    int mode, int relu, int Trows, int TT, int toff)
{
    __shared__ unsigned short sA[2][128 * 32];
    __shared__ unsigned short sB[2][128 * 32];
    int tid = threadIdx.x, wv = tid >> 5, lane = tid & 31;
    int wm = wv & 3, wn = wv >> 2;
    int m0 = blockIdx.x * 128, n0 = blockIdx.y * 128;
    int l16 = lane & 15;
    int half8  = (lane < 16) ? 0 : 8;
    int half16 = (lane < 16) ? 0 : 16;

    f32x8 c[2][4];
#pragma unroll
    for (int f = 0; f < 2; f++)
#pragma unroll
        for (int g = 0; g < 4; g++) c[f][g] = zero8();

    auto stage = [&](int buf, int k0) {
#pragma unroll
        for (int i = 0; i < 2; i++) {          // 512 x 16B chunks per tile
            int lin = tid + i * 256;
            int row = lin >> 2, c4 = lin & 3;
            cp16_async(&sA[buf][row * 32 + c4 * 8],
                       A + (size_t)(m0 + row) * K + k0 + c4 * 8);
            cp16_async(&sB[buf][row * 32 + c4 * 8],
                       W + (size_t)(n0 + row) * K + k0 + c4 * 8);
        }
    };

    int nk = K >> 5;
    stage(0, 0);
    for (int ki = 0; ki < nk; ki++) {
        wait_async0();           // this wave's async tile complete
        __syncthreads();         // all waves' tiles visible; prev compute done
        int cur = ki & 1;
        if (ki + 1 < nk) stage(cur ^ 1, (ki + 1) << 5);   // prefetch next tile async

        Frag a[2], b[4];
#pragma unroll
        for (int f = 0; f < 2; f++) {          // A 16x32: interleaved K halves
            int r = wm * 32 + f * 16 + l16;
            a[f].u[0] = *(const u32x4*)(&sA[cur][r * 32 + half8]);
            a[f].u[1] = *(const u32x4*)(&sA[cur][r * 32 + 16 + half8]);
        }
#pragma unroll
        for (int g = 0; g < 4; g++) {          // B 32x16: interleaved K halves
            int col = wn * 64 + g * 16 + l16;
            b[g].u[0] = *(const u32x4*)(&sB[cur][col * 32 + half8]);
            b[g].u[1] = *(const u32x4*)(&sB[cur][col * 32 + 16 + half8]);
        }
#pragma unroll
        for (int f = 0; f < 2; f++)
#pragma unroll
            for (int g = 0; g < 4; g++)
                c[f][g] = wmma_bf(a[f], b[g], c[f][g]);
    }

    // epilogue
#pragma unroll
    for (int f = 0; f < 2; f++)
#pragma unroll
        for (int g = 0; g < 4; g++)
#pragma unroll
            for (int r = 0; r < 8; r++) {
                int m = m0 + wm * 32 + f * 16 + r + ((lane < 16) ? 0 : 8);
                int n = n0 + wn * 64 + g * 16 + l16;
                float v = c[f][g][r];
                if (bias) v += bias[n];
                if (relu) v = fmaxf(v, 0.f);
                if (mode == 1) {
                    float rv = resid ? resid[(size_t)m * N + n] : 0.f;
                    ((float*)out)[(size_t)m * N + n] = v + rv;
                } else if (mode == 0) {
                    int bb = m / Trows, t = m % Trows + toff;
                    ((unsigned short*)out)[((size_t)bb * TT + t) * N + n] = f2bf(v);
                } else {
                    int bb = m / Trows, t = m % Trows + toff;
                    int h = n / HDIM, d = n % HDIM;
                    ((unsigned short*)out)[(((size_t)bb * NHEAD + h) * HDIM + d) * TT + t] = f2bf(v);
                }
            }
}

// =====================================================================
// Flash attention: one WG (128 thr = 4 waves) per (b,h).
// Qn  bf16 [B,H,64,96]  (scale folded in)
// Kn  bf16 [B,H,2368,96]
// Vt  bf16 [B,H,96,2368] (d-major)
// out bf16 [B,64,1536]
// TJ = 64 (2368 = 37*64 exactly).  K/V tiles staged via async->LDS.
// =====================================================================
__global__ __launch_bounds__(128) void attn_kernel(
    const unsigned short* __restrict__ Qn, const unsigned short* __restrict__ Kn,
    const unsigned short* __restrict__ Vt, unsigned short* __restrict__ out)
{
    __shared__ unsigned short sK[64 * 96];   // [t][d]
    __shared__ unsigned short sV[96 * 64];   // [d][t]
    __shared__ unsigned short sP[4][16 * 64];

    int bh = blockIdx.x;
    int b  = bh >> 4, h = bh & 15;
    int tid = threadIdx.x, wv = tid >> 5, lane = tid & 31;
    int l16 = lane & 15;
    int half8  = (lane < 16) ? 0 : 8;
    int half16 = (lane < 16) ? 0 : 16;

    // Q fragments for this wave's 16 query rows, held in registers (k = 96 -> 3 frags)
    const unsigned short* qptr = Qn + (size_t)bh * NL * HDIM;
    Frag qa[3];
#pragma unroll
    for (int kk = 0; kk < 3; kk++) {
        const unsigned short* src = qptr + (size_t)(wv * 16 + l16) * HDIM + kk * 32 + half8;
        qa[kk].u[0] = *(const u32x4*)src;
        qa[kk].u[1] = *(const u32x4*)(src + 16);
    }

    f32x8 acc[6];
#pragma unroll
    for (int g = 0; g < 6; g++) acc[g] = zero8();
    float mrow[8], lrow[8];
#pragma unroll
    for (int r = 0; r < 8; r++) { mrow[r] = -INFINITY; lrow[r] = 0.f; }

    const unsigned short* kbase = Kn + (size_t)bh * TTOT * HDIM;
    const unsigned short* vbase = Vt + (size_t)bh * HDIM * TTOT;

    for (int j0 = 0; j0 < TTOT; j0 += 64) {
        __syncthreads();
        // stage K tile [64][96] and V tile [96][64] via async global->LDS
        for (int c2 = tid; c2 < 768; c2 += 128) {
            int rowt = c2 / 12, c8 = c2 % 12;
            cp16_async(&sK[rowt * 96 + c8 * 8],
                       kbase + (size_t)(j0 + rowt) * HDIM + c8 * 8);
        }
        for (int c2 = tid; c2 < 768; c2 += 128) {
            int rowd = c2 >> 3, c8 = c2 & 7;
            cp16_async(&sV[rowd * 64 + c8 * 8],
                       vbase + (size_t)rowd * TTOT + j0 + c8 * 8);
        }
        wait_async0();
        __syncthreads();

        // scores 16x64 = Q(16x96) * K^T(96x64)
        f32x8 sc[4];
#pragma unroll
        for (int nf = 0; nf < 4; nf++) {
            f32x8 cc = zero8();
#pragma unroll
            for (int kk = 0; kk < 3; kk++) {
                Frag bk;
                const unsigned short* src = sK + (nf * 16 + l16) * 96 + kk * 32 + half8;
                bk.u[0] = *(const u32x4*)src;
                bk.u[1] = *(const u32x4*)(src + 16);
                cc = wmma_bf(qa[kk], bk, cc);
            }
            sc[nf] = cc;
        }

        // online softmax per row (16-lane butterfly halves == row halves of C layout)
#pragma unroll
        for (int r = 0; r < 8; r++) {
            float v = fmaxf(fmaxf(sc[0][r], sc[1][r]), fmaxf(sc[2][r], sc[3][r]));
#pragma unroll
            for (int o = 1; o < 16; o <<= 1) v = fmaxf(v, __shfl_xor(v, o, 32));
            float mn = fmaxf(mrow[r], v);
            float corr = __expf(mrow[r] - mn);
            mrow[r] = mn;
            float sum = 0.f;
#pragma unroll
            for (int nf = 0; nf < 4; nf++) { sc[nf][r] = __expf(sc[nf][r] - mn); sum += sc[nf][r]; }
#pragma unroll
            for (int o = 1; o < 16; o <<= 1) sum += __shfl_xor(sum, o, 32);
            lrow[r] = lrow[r] * corr + sum;
#pragma unroll
            for (int g = 0; g < 6; g++) acc[g][r] *= corr;
        }

        // stage P (bf16) for WMMA A-frag reload (per-wave region, no barrier needed)
#pragma unroll
        for (int nf = 0; nf < 4; nf++)
#pragma unroll
            for (int r = 0; r < 8; r++) {
                int rowp = r + ((lane < 16) ? 0 : 8);
                sP[wv][rowp * 64 + nf * 16 + l16] = f2bf(sc[nf][r]);
            }

        // acc += P(16x64) * V(64x96)
#pragma unroll
        for (int kt = 0; kt < 2; kt++) {
            Frag pa;
            const unsigned short* src = &sP[wv][l16 * 64 + kt * 32 + half8];
            pa.u[0] = *(const u32x4*)src;
            pa.u[1] = *(const u32x4*)(src + 16);
#pragma unroll
            for (int g = 0; g < 6; g++) {
                Frag bv;
                const unsigned short* sv = sV + (g * 16 + l16) * 64 + kt * 32 + half8;
                bv.u[0] = *(const u32x4*)sv;
                bv.u[1] = *(const u32x4*)(sv + 16);
                acc[g] = wmma_bf(pa, bv, acc[g]);
            }
        }
    }

    // normalize and write [B,64,1536]
#pragma unroll
    for (int g = 0; g < 6; g++)
#pragma unroll
        for (int r = 0; r < 8; r++) {
            int rowq = wv * 16 + r + ((lane < 16) ? 0 : 8);
            int d = g * 16 + l16;
            float v = acc[g][r] / lrow[r];
            out[((size_t)(b * NL + rowq)) * NHHD + h * HDIM + d] = f2bf(v);
        }
}

// =====================================================================
// Host launch
// =====================================================================
extern "C" void kernel_launch(void* const* d_in, const int* in_sizes, int n_in,
                              void* d_out, int out_size, void* d_ws, size_t ws_size,
                              hipStream_t stream)
{
    const float* context = (const float*)d_in[0];
    const float* latents = (const float*)d_in[1];
    const float* ctx_g = (const float*)d_in[2];
    const float* ctx_b = (const float*)d_in[3];
    const float* lat_g = (const float*)d_in[4];
    const float* lat_b = (const float*)d_in[5];
    const float* q_g   = (const float*)d_in[6];
    const float* q_b   = (const float*)d_in[7];
    const float* k_g   = (const float*)d_in[8];
    const float* k_b   = (const float*)d_in[9];
    const float* Wq    = (const float*)d_in[10];
    const float* Wk    = (const float*)d_in[11];
    const float* Wv    = (const float*)d_in[12];
    const float* Wo    = (const float*)d_in[13];
    const float* mlp_g = (const float*)d_in[14];
    const float* mlp_b = (const float*)d_in[15];
    const float* Wfc   = (const float*)d_in[16];
    const float* Wcp   = (const float*)d_in[17];
    const float* f_g   = (const float*)d_in[18];
    const float* f_b   = (const float*)d_in[19];
    (void)in_sizes; (void)n_in; (void)out_size; (void)ws_size;

    const int M_CTX = BBATCH * SSEQ;   // 73728
    const int M_LAT = BBATCH * NL;     // 2048

    // ---- workspace carve ----
    char* p = (char*)d_ws;
    auto alloc = [&](size_t bytes) -> void* {
        void* r = (void*)p;
        p += (bytes + 255) & ~(size_t)255;
        return r;
    };
    unsigned short* Xcn  = (unsigned short*)alloc((size_t)M_CTX * DD * 2);      // normalized ctx, bf16
    float*          lat  = (float*)alloc((size_t)M_LAT * DD * 4);               // latent state f32
    unsigned short* Ln   = (unsigned short*)alloc((size_t)M_LAT * DD * 2);      // LN(latents)
    unsigned short* Hn   = (unsigned short*)alloc((size_t)M_LAT * DD * 2);      // LN pre-MLP
    unsigned short* Qraw = (unsigned short*)alloc((size_t)M_LAT * NHHD * 2);
    unsigned short* Qn   = (unsigned short*)alloc((size_t)M_LAT * NHHD * 2);
    unsigned short* AttO = (unsigned short*)alloc((size_t)M_LAT * NHHD * 2);
    unsigned short* Ffc  = (unsigned short*)alloc((size_t)M_LAT * INTERN * 2);
    unsigned short* buf1 = (unsigned short*)alloc((size_t)BBATCH * TTOT * NHHD * 2); // Kraw then Vh
    unsigned short* Kn   = (unsigned short*)alloc((size_t)BBATCH * TTOT * NHHD * 2);
    unsigned short* WqT  = (unsigned short*)alloc((size_t)NHHD * DD * 2);
    unsigned short* WkTc = (unsigned short*)alloc((size_t)NHHD * DD * 2);
    unsigned short* WkTp = (unsigned short*)alloc((size_t)NHHD * DD * 2);
    unsigned short* WvTc = (unsigned short*)alloc((size_t)NHHD * DD * 2);
    unsigned short* WvTp = (unsigned short*)alloc((size_t)NHHD * DD * 2);
    unsigned short* WoT  = (unsigned short*)alloc((size_t)DD * NHHD * 2);
    unsigned short* WfcT = (unsigned short*)alloc((size_t)INTERN * DD * 2);
    unsigned short* WcpT = (unsigned short*)alloc((size_t)DD * INTERN * 2);
    float* bkc = (float*)alloc(NHHD * 4);
    float* bvc = (float*)alloc(NHHD * 4);

    auto gemm = [&](const unsigned short* A, const unsigned short* W, const float* bias,
                    const float* resid, void* out, int M, int N, int K,
                    int mode, int relu, int Trows, int TT, int toff) {
        dim3 g(M / 128, N / 128);
        gemm_bf16_kernel<<<g, 256, 0, stream>>>(A, W, bias, resid, out, M, N, K,
                                                mode, relu, Trows, TT, toff);
    };
    auto tsp = [&](const float* W, const float* g, unsigned short* WT, int K, int N) {
        tsp_kernel<<<dim3(N / 32, K / 32), dim3(32, 8), 0, stream>>>(W, g, WT, K, N);
    };

    const float qscale = 1.0f / sqrtf((float)HDIM);

    // ---- one-time: normalize context (no g/b; folded into weights), init latent state ----
    ln_rows_kernel<<<(M_CTX + 7) / 8, 256, 0, stream>>>(context, nullptr, nullptr, Xcn, M_CTX, DD, 0);
    bcast_lat_kernel<<<(M_LAT * DD + 255) / 256, 256, 0, stream>>>(latents, lat, M_LAT * DD, NL * DD);

    for (int i = 0; i < 6; i++) {
        const float* Wq_i  = Wq  + (size_t)i * DD * NHHD;
        const float* Wk_i  = Wk  + (size_t)i * DD * NHHD;
        const float* Wv_i  = Wv  + (size_t)i * DD * NHHD;
        const float* Wo_i  = Wo  + (size_t)i * NHHD * DD;
        const float* Wfc_i = Wfc + (size_t)i * DD * INTERN;
        const float* Wcp_i = Wcp + (size_t)i * INTERN * DD;

        // ---- weight prep: transposed bf16 copies; ctx-LN gamma folded, beta -> bias row ----
        tsp(Wq_i,  nullptr,          WqT,  DD, NHHD);
        tsp(Wk_i,  ctx_g + i * DD,   WkTc, DD, NHHD);
        tsp(Wk_i,  nullptr,          WkTp, DD, NHHD);
        tsp(Wv_i,  ctx_g + i * DD,   WvTc, DD, NHHD);
        tsp(Wv_i,  nullptr,          WvTp, DD, NHHD);
        tsp(Wo_i,  nullptr,          WoT,  NHHD, DD);
        tsp(Wfc_i, nullptr,          WfcT, DD, INTERN);
        tsp(Wcp_i, nullptr,          WcpT, INTERN, DD);
        gemv_bias_kernel<<<NHHD / 256, 256, 0, stream>>>(ctx_b + i * DD, Wk_i, bkc, DD, NHHD);
        gemv_bias_kernel<<<NHHD / 256, 256, 0, stream>>>(ctx_b + i * DD, Wv_i, bvc, DD, NHHD);

        // ---- latent LN (explicit) ----
        ln_rows_kernel<<<(M_LAT + 7) / 8, 256, 0, stream>>>(lat, lat_g + i * DD, lat_b + i * DD,
                                                            Ln, M_LAT, DD, 0);

        // ---- K projection (ctx rows folded-LN + latent rows) -> Kraw=buf1 [B,2368,1536] ----
        gemm(Xcn, WkTc, bkc,     nullptr, buf1, M_CTX, NHHD, DD, 0, 0, SSEQ, TTOT, 0);
        gemm(Ln,  WkTp, nullptr, nullptr, buf1, M_LAT, NHHD, DD, 0, 0, NL,   TTOT, SSEQ);
        // per-head k-LN -> Kn [B,H,2368,96]
        ln_head_kernel<<<((size_t)BBATCH * TTOT * NHEAD + 7) / 8, 256, 0, stream>>>(
            buf1, k_g + i * HDIM, k_b + i * HDIM, Kn, BBATCH * TTOT, TTOT, TTOT, 0, 1.0f);

        // ---- V projection -> Vh = buf1 (d-major [B,H,96,2368]); Kraw dead now ----
        gemm(Xcn, WvTc, bvc,     nullptr, buf1, M_CTX, NHHD, DD, 2, 0, SSEQ, TTOT, 0);
        gemm(Ln,  WvTp, nullptr, nullptr, buf1, M_LAT, NHHD, DD, 2, 0, NL,   TTOT, SSEQ);

        // ---- Q projection + per-head q-LN (HD^-0.5 folded in) ----
        gemm(Ln, WqT, nullptr, nullptr, Qraw, M_LAT, NHHD, DD, 0, 0, M_LAT, M_LAT, 0);
        ln_head_kernel<<<((size_t)M_LAT * NHEAD + 7) / 8, 256, 0, stream>>>(
            Qraw, q_g + i * HDIM, q_b + i * HDIM, Qn, M_LAT, NL, NL, 0, qscale);

        // ---- flash attention ----
        attn_kernel<<<BBATCH * NHEAD, 128, 0, stream>>>(Qn, Kn, buf1, AttO);

        // ---- output projection + residual into lat ----
        gemm(AttO, WoT, nullptr, lat, lat, M_LAT, DD, NHHD, 1, 0, M_LAT, M_LAT, 0);

        // ---- MLP ----
        ln_rows_kernel<<<(M_LAT + 7) / 8, 256, 0, stream>>>(lat, mlp_g + i * DD, mlp_b + i * DD,
                                                            Hn, M_LAT, DD, 0);
        gemm(Hn,  WfcT, nullptr, nullptr, Ffc, M_LAT, INTERN, DD,     0, 1, M_LAT, M_LAT, 0);
        gemm(Ffc, WcpT, nullptr, lat,     lat, M_LAT, DD,     INTERN, 1, 0, M_LAT, M_LAT, 0);
    }

    // ---- final LN -> f32 output [B,64,1280] ----
    ln_rows_kernel<<<(M_LAT + 7) / 8, 256, 0, stream>>>(lat, f_g, f_b, d_out, M_LAT, DD, 1);
}
